// SimpleInfoNCE_369367187549
// MI455X (gfx1250) — matrix operand order repaired and checked by reading, not compiled, not run on hardware
//
#include <hip/hip_runtime.h>

typedef _Float16 v16h __attribute__((ext_vector_type(16)));
typedef float    v8f  __attribute__((ext_vector_type(8)));
typedef float    v4f  __attribute__((ext_vector_type(4)));

#define N_ROWS 8192
#define DIM    256
#define LDS_STRIDE 264   // halves per staged row (132 dwords -> 4-bank rotation per row)

// ---------------- Kernel 1: row L2-normalize, fp32 -> f16 ----------------
__global__ __launch_bounds__(256) void infonce_normalize(const float* __restrict__ F,
                                                         _Float16* __restrict__ Fh) {
  __shared__ float red[256];
  const int row = blockIdx.x;
  const int t = threadIdx.x;
  float x = F[(size_t)row * DIM + t];
  red[t] = x * x;
  __syncthreads();
  for (int s = 128; s > 0; s >>= 1) {
    if (t < s) red[t] += red[t + s];
    __syncthreads();
  }
  const float inv = rsqrtf(red[0]);
  Fh[(size_t)row * DIM + t] = (_Float16)(x * inv);
}

// ---------------- Kernel 2: tiled F F^T via WMMA + streaming exp-sums ----------------
// grid.x = 64 row-groups (8 row-blocks of 16 rows, one per wave)
// grid.y = 8  column slices (64 column-blocks of 16; processed 2 tiles per barrier pair)
__global__ __launch_bounds__(256) void infonce_main(const _Float16* __restrict__ Fh,
                                                    float* __restrict__ allPart,
                                                    float* __restrict__ posPart) {
  __shared__ __align__(32) _Float16 colT[32 * LDS_STRIDE];   // two 16-col tiles, ~16.5 KB

  const int t    = threadIdx.x;
  const int lane = t & 31;
  const int wave = t >> 5;
  const int rb   = blockIdx.x * 8 + wave;   // row block 0..511 (rows 16*rb .. +15)
  const int cs   = blockIdx.y;              // column slice: col blocks [cs*64, cs*64+64)

  const int hi = lane >> 4;                 // half-wave select
  const int ln = lane & 15;

  // ---- A fragments: 16 rows x 256 K, register resident (ISA 16-bit A layout) ----
  // lane (m=ln, hi): VGPR v holds K pair (v<4?0:16)+(v&3)*2+hi*8 (+kb*32)
  // => 16 halves = two contiguous 8-half runs at k0 = kb*32 + hi*8 and k0+16.
  v16h afrag[8];
  {
    const _Float16* ap = Fh + (size_t)(rb * 16 + ln) * DIM;
#pragma unroll
    for (int kb = 0; kb < 8; ++kb) {
      union { v4f f[2]; v16h v; } u;
      u.f[0] = *(const v4f*)(ap + kb * 32 + hi * 8);
      u.f[1] = *(const v4f*)(ap + kb * 32 + 16 + hi * 8);
      afrag[kb] = u.v;
    }
  }

  v8f accA = {};   // sum_j exp(sim-2), j != i
  v8f accP = {};   // positives only

  // cooperative stage role: 8 threads per staged row (32 rows), 32 halves (64B) each
  const int grow   = t >> 3;          // 0..31
  const int gchunk = (t & 7) * 32;    // half offset within row
  const _Float16* gptr = Fh + (size_t)((cs * 64) * 16 + grow) * DIM + gchunk;
  const size_t gstep = (size_t)32 * DIM;   // halves per pair-iteration

  // prime the register prefetch pipeline
  v4f pre0 = ((const v4f*)gptr)[0];
  v4f pre1 = ((const v4f*)gptr)[1];
  v4f pre2 = ((const v4f*)gptr)[2];
  v4f pre3 = ((const v4f*)gptr)[3];

  for (int ci2 = 0; ci2 < 32; ++ci2) {
    __syncthreads();                        // previous compute done: safe to overwrite LDS
    {
      v4f* s = (v4f*)(colT + grow * LDS_STRIDE + gchunk);
      s[0] = pre0; s[1] = pre1; s[2] = pre2; s[3] = pre3;
    }
    __syncthreads();                        // staged tiles visible

    // prefetch next pair of tiles; loads overlap the 16 WMMAs below
    if (ci2 + 1 < 32) {
      const v4f* g = (const v4f*)(gptr + (size_t)(ci2 + 1) * gstep);
      pre0 = g[0]; pre1 = g[1]; pre2 = g[2]; pre3 = g[3];
    }

    // ---- two 16x16 tiles of F F^T, K=256: two independent 8-deep WMMA chains ----
    const int c0 = cs * 64 + 2 * ci2;
    const int c1 = c0 + 1;
    v8f d0 = {}, d1 = {};
#pragma unroll
    for (int kb = 0; kb < 8; ++kb) {
      // B fragment (same K order as A): lane n holds two 8-half runs at
      // kb*32 + hi*8 and kb*32 + 16 + hi*8, for column row n of the staged tile.
      const _Float16* bp0 = colT + ln * LDS_STRIDE + kb * 32 + hi * 8;
      const _Float16* bp1 = bp0 + 16 * LDS_STRIDE;
      union { v4f f[2]; v16h v; } u0, u1;
      u0.f[0] = *(const v4f*)(bp0); u0.f[1] = *(const v4f*)(bp0 + 16);
      u1.f[0] = *(const v4f*)(bp1); u1.f[1] = *(const v4f*)(bp1 + 16);
      d0 = __builtin_amdgcn_wmma_f32_16x16x32_f16(
             false, afrag[kb], false, u0.v, (short)0, d0, false, false);
      d1 = __builtin_amdgcn_wmma_f32_16x16x32_f16(
             false, afrag[kb], false, u1.v, (short)0, d1, false, false);
    }

    // sim = dot/T = 2*dot ; stable shift by the max possible value 2 (rows normalized).
    auto accumulate = [&](const v8f& d, int c) {
      if (c != rb) {
#pragma unroll
        for (int r = 0; r < 8; ++r) accA[r] += __expf(2.0f * d[r] - 2.0f);
      } else {
        const int mbase = hi * 8;           // D rows r+mbase live in this half
#pragma unroll
        for (int r = 0; r < 8; ++r) {
          const int m = r + mbase;          // local row within tile
          const float e = __expf(2.0f * d[r] - 2.0f);
          if (m != ln) {                    // exclude self
            accA[r] += e;
            if ((m >> 2) == (ln >> 2)) accP[r] += e;   // same group of k=4
          }
        }
      }
    };
    accumulate(d0, c0);
    accumulate(d1, c1);
  }

  // ---- reduce across the 16 lanes of each half (per-row sums) ----
#pragma unroll
  for (int r = 0; r < 8; ++r) {
    float a = accA[r], p = accP[r];
    for (int off = 1; off < 16; off <<= 1) {
      a += __shfl_xor(a, off, 32);
      p += __shfl_xor(p, off, 32);
    }
    accA[r] = a; accP[r] = p;
  }

  // lanes ln==0 of each half hold the 8 row totals (rows hi*8 + r)
  if (ln == 0) {
    const int rowbase = rb * 16 + hi * 8;
#pragma unroll
    for (int r = 0; r < 8; ++r) {
      allPart[(size_t)cs * N_ROWS + rowbase + r] = accA[r];
      posPart[(size_t)cs * N_ROWS + rowbase + r] = accP[r];
    }
  }
}

// ---------------- Kernel 3: combine slices, logs, deterministic mean ----------------
__global__ __launch_bounds__(1024) void infonce_finalize(const float* __restrict__ allPart,
                                                         const float* __restrict__ posPart,
                                                         float* __restrict__ out) {
  __shared__ float red[1024];
  const int t = threadIdx.x;
  float loss = 0.0f;
  for (int row = t; row < N_ROWS; row += 1024) {
    float sa = 0.0f, sp = 0.0f;
#pragma unroll
    for (int s = 0; s < 8; ++s) {
      sa += allPart[(size_t)s * N_ROWS + row];
      sp += posPart[(size_t)s * N_ROWS + row];
    }
    // loss_i = lse_all - lse_pos (the constant shift cancels)
    loss += __logf(sa) - __logf(sp);
  }
  red[t] = loss;
  __syncthreads();
  for (int s = 512; s > 0; s >>= 1) {
    if (t < s) red[t] += red[t + s];
    __syncthreads();
  }
  if (t == 0) out[0] = red[0] / (float)N_ROWS;
}

extern "C" void kernel_launch(void* const* d_in, const int* in_sizes, int n_in,
                              void* d_out, int out_size, void* d_ws, size_t ws_size,
                              hipStream_t stream) {
  (void)in_sizes; (void)n_in; (void)out_size; (void)ws_size;
  const float* F = (const float*)d_in[0];   // [8192, 256] fp32

  // workspace: f16 features (4MB) | allPart (8*8192 f32) | posPart (8*8192 f32)
  _Float16* Fh   = (_Float16*)d_ws;
  float* allPart = (float*)((char*)d_ws + (size_t)N_ROWS * DIM * sizeof(_Float16));
  float* posPart = allPart + (size_t)8 * N_ROWS;
  float* out     = (float*)d_out;

  infonce_normalize<<<N_ROWS, 256, 0, stream>>>(F, Fh);
  infonce_main<<<dim3(64, 8), 256, 0, stream>>>(Fh, allPart, posPart);
  infonce_finalize<<<1, 1024, 0, stream>>>(allPart, posPart, out);
}
